// VanillaMasker_10668698763375
// MI455X (gfx1250) — matrix operand order repaired and checked by reading, not compiled, not run on hardware
//
#include <hip/hip_runtime.h>
#include <hip/hip_bf16.h>

// ---------------- problem constants ----------------
#define BATCH   16
#define C_IN    256
#define C_HID   128
#define HW      64
#define HP      66            // padded spatial (1-px halo)
#define NPOS    4096
#define NTOK    4096
#define TOPK    2048
#define LBL     19
#define GRPS    32
#define CODE    32
#define EPSV    1e-5f

typedef __bf16 bf16_t;
typedef bf16_t          v16bf __attribute__((ext_vector_type(16)));
typedef float           v8f   __attribute__((ext_vector_type(8)));
typedef unsigned short  v8us  __attribute__((ext_vector_type(8)));

// ---------------- helpers ----------------
__device__ __forceinline__ unsigned short f2bf(float f) {
    unsigned int u = __float_as_uint(f);
    unsigned int r = u + 0x7FFFu + ((u >> 16) & 1u);   // round-to-nearest-even
    return (unsigned short)(r >> 16);
}

// ---------------- weight repack: tap-major K, A-fragment order ----------------
// K index kk = tap*CINp + cin  (CINp = padded input channels, power of 2, mult of 32)
// Apack[ct][kt][lane][i] -> each lane's 16 bf16 contiguous (32B) = one vector load.
// A layout (16x32 bf16): lanes 0-15 K={0..15}, lanes 16-31 K={16..31}, M=lane&15
__global__ void cvt_weight_frag(const float* __restrict__ w, unsigned short* __restrict__ Ap,
                                int Cout, int CinReal, int CINp, int ntaps) {
    int nkt = (ntaps * CINp) >> 5;
    int total = (Cout >> 4) * nkt * 512;
    int idx = blockIdx.x * blockDim.x + threadIdx.x;
    if (idx >= total) return;
    int i    = idx & 15;
    int lane = (idx >> 4) & 31;
    int rest = idx >> 9;
    int kt   = rest % nkt;
    int ct   = rest / nkt;
    int hi = lane >> 4, m = lane & 15;
    int kk  = kt * 32 + hi * 16 + i;
    int tap = kk / CINp;
    int cin = kk - tap * CINp;
    int c   = ct * 16 + m;
    float v = (cin < CinReal) ? w[(size_t)c * CinReal * ntaps + cin * ntaps + tap] : 0.f;
    Ap[idx] = f2bf(v);
}

// ---------------- semantic -> zero-padded NHWC bf16 [B][66][66][32] ----------------
__global__ void cvt_seg_pad(const float* __restrict__ sem, unsigned short* __restrict__ segP) {
    int idx = blockIdx.x * blockDim.x + threadIdx.x;
    int total = BATCH * HP * HP * 32;
    if (idx >= total) return;
    int cin  = idx & 31;
    int rest = idx >> 5;
    int xx = rest % HP; rest /= HP;
    int yy = rest % HP;
    int b  = rest / HP;
    float v = 0.f;
    if (cin < LBL && yy >= 1 && yy <= HW && xx >= 1 && xx <= HW)
        v = sem[((size_t)(b * LBL + cin) * HW + (yy - 1)) * HW + (xx - 1)];
    segP[idx] = f2bf(v);
}

// ---------------- group-norm stats (C=256, 32 groups of 8 ch) ----------------
__global__ void __launch_bounds__(256)
gn_stats(const float* __restrict__ x, float* __restrict__ mean, float* __restrict__ var) {
    int b = blockIdx.x, g = blockIdx.y;
    const float* base = x + ((size_t)b * C_IN + g * 8) * NPOS;
    const int n = 8 * NPOS;
    float s = 0.f, q = 0.f;
    for (int i = threadIdx.x; i < n; i += 256) { float v = base[i]; s += v; q += v * v; }
    __shared__ float ls[256], lq[256];
    ls[threadIdx.x] = s; lq[threadIdx.x] = q;
    __syncthreads();
    for (int o = 128; o > 0; o >>= 1) {
        if (threadIdx.x < o) { ls[threadIdx.x] += ls[threadIdx.x + o]; lq[threadIdx.x] += lq[threadIdx.x + o]; }
        __syncthreads();
    }
    if (threadIdx.x == 0) {
        float m = ls[0] / n;
        mean[b * GRPS + g] = m;
        var [b * GRPS + g] = lq[0] / n - m * m;
    }
}

// ---------------- 3x3 conv, implicit GEMM, tap-outer / channel-tile-inner ----------------
// input: zero-padded NHWC bf16 [B][66][66][CINP]; weights: prepacked tap-major fragments.
// Inner loop fully unrolled with compile-time CINP -> all offsets become immediates.
template <int CINP>
__global__ void __launch_bounds__(32)
conv3x3_wmma(const unsigned short* __restrict__ xp,
             const unsigned short* __restrict__ Ap,
             const float* __restrict__ bias, int Cout,
             unsigned short* __restrict__ outP,    // NHWC padded bf16 [B][66][66][Cout], optional
             float* __restrict__ outF32,           // NCHW fp32, optional
             int relu) {
    constexpr int NCI = CINP / 32;                 // channel tiles per tap
    constexpr int NKT = 9 * NCI;
    const int lane = threadIdx.x;
    const int ct  = blockIdx.x;
    const int pt2 = blockIdx.y;                    // row y = pt2>>1, x0 = (pt2&1)*32
    const int b   = blockIdx.z;
    const int y   = pt2 >> 1;
    const int x0  = (pt2 & 1) * 32;
    const int n   = lane & 15;
    const int hi  = lane >> 4;
    const int col0 = x0 + n, col1 = x0 + 16 + n;

    v8f acc0 = {}, acc1 = {};
    const unsigned short* abase = Ap + (((size_t)ct * NKT * 32) + lane) * 16;

    for (int tap = 0; tap < 9; ++tap) {
        int t3 = tap / 3;
        int dy = t3 - 1, dx = tap - t3 * 3 - 1;
        const unsigned short* pa = abase + (size_t)tap * NCI * 512;
        const unsigned short* p0 =
            xp + (((size_t)b * HP + (y + 1 + dy)) * HP + (col0 + 1 + dx)) * CINP + hi * 16;
        const unsigned short* p1 =
            xp + (((size_t)b * HP + (y + 1 + dy)) * HP + (col1 + 1 + dx)) * CINP + hi * 16;
        if (tap < 8) __builtin_prefetch(pa + NCI * 512, 0, 1);
#pragma unroll
        for (int ci = 0; ci < NCI; ++ci) {
            v16bf a  = *reinterpret_cast<const v16bf*>(pa + ci * 512);
            v16bf b0 = *reinterpret_cast<const v16bf*>(p0 + ci * 32);
            v16bf b1 = *reinterpret_cast<const v16bf*>(p1 + ci * 32);
            acc0 = __builtin_amdgcn_wmma_f32_16x16x32_bf16(false, a, false, b0, (short)0, acc0, false, false);
            acc1 = __builtin_amdgcn_wmma_f32_16x16x32_bf16(false, a, false, b1, (short)0, acc1, false, false);
        }
    }

#pragma unroll
    for (int t = 0; t < 2; ++t) {
        int col = x0 + t * 16 + n;
        v8us pk;
#pragma unroll
        for (int r = 0; r < 8; ++r) {
            int ch = ct * 16 + r + hi * 8;
            float v = (t ? acc1[r] : acc0[r]) + bias[ch];
            if (relu) v = fmaxf(v, 0.f);
            pk[r] = f2bf(v);
            if (outF32)
                outF32[((size_t)b * Cout + ch) * NPOS + y * HW + col] = v;
        }
        if (outP) {
            unsigned short* dst =
                outP + (((size_t)b * HP + (y + 1)) * HP + (col + 1)) * Cout + ct * 16 + hi * 8;
            *reinterpret_cast<v8us*>(dst) = pk;   // packed 16B NHWC store
        }
    }
}

// ---------------- fused SPADE: gamma & beta convs share B tiles; GN applied in epilogue ----
__global__ void __launch_bounds__(32)
spade_gb_apply(const unsigned short* __restrict__ actvP, // NHWC padded [B][66][66][128]
               const unsigned short* __restrict__ Apg,
               const unsigned short* __restrict__ Apb,
               const float* __restrict__ bg, const float* __restrict__ bbias,
               const float* __restrict__ x,              // NCHW fp32 (tensor being normalized)
               const float* __restrict__ gmean, const float* __restrict__ gvar,
               unsigned short* __restrict__ outP,        // NHWC padded [B][66][66][256], optional
               float* __restrict__ scoreAcc,             // optional 1x1-conv partial sums
               const float* __restrict__ c2w) {
    constexpr int CINP = C_HID;
    constexpr int NCI  = CINP / 32;                      // 4
    constexpr int NKT  = 9 * NCI;                        // 36
    const int Cout = C_IN;
    const int lane = threadIdx.x;
    const int ct = blockIdx.x, pt2 = blockIdx.y, b = blockIdx.z;
    const int y  = pt2 >> 1;
    const int x0 = (pt2 & 1) * 32;
    const int n  = lane & 15;
    const int hi = lane >> 4;
    const int col0 = x0 + n, col1 = x0 + 16 + n;

    v8f accg[2] = {}, accb[2] = {};
    const unsigned short* gbase = Apg + (((size_t)ct * NKT * 32) + lane) * 16;
    const unsigned short* bbase = Apb + (((size_t)ct * NKT * 32) + lane) * 16;

    for (int tap = 0; tap < 9; ++tap) {
        int t3 = tap / 3;
        int dy = t3 - 1, dx = tap - t3 * 3 - 1;
        const unsigned short* pg = gbase + (size_t)tap * NCI * 512;
        const unsigned short* pb = bbase + (size_t)tap * NCI * 512;
        const unsigned short* p0 =
            actvP + (((size_t)b * HP + (y + 1 + dy)) * HP + (col0 + 1 + dx)) * CINP + hi * 16;
        const unsigned short* p1 =
            actvP + (((size_t)b * HP + (y + 1 + dy)) * HP + (col1 + 1 + dx)) * CINP + hi * 16;
        if (tap < 8) {
            __builtin_prefetch(pg + NCI * 512, 0, 1);
            __builtin_prefetch(pb + NCI * 512, 0, 1);
        }
#pragma unroll
        for (int ci = 0; ci < NCI; ++ci) {
            v16bf ag = *reinterpret_cast<const v16bf*>(pg + ci * 512);
            v16bf ab = *reinterpret_cast<const v16bf*>(pb + ci * 512);
            v16bf b0 = *reinterpret_cast<const v16bf*>(p0 + ci * 32);
            v16bf b1 = *reinterpret_cast<const v16bf*>(p1 + ci * 32);
            accg[0] = __builtin_amdgcn_wmma_f32_16x16x32_bf16(false, ag, false, b0, (short)0, accg[0], false, false);
            accb[0] = __builtin_amdgcn_wmma_f32_16x16x32_bf16(false, ab, false, b0, (short)0, accb[0], false, false);
            accg[1] = __builtin_amdgcn_wmma_f32_16x16x32_bf16(false, ag, false, b1, (short)0, accg[1], false, false);
            accb[1] = __builtin_amdgcn_wmma_f32_16x16x32_bf16(false, ab, false, b1, (short)0, accb[1], false, false);
        }
    }

#pragma unroll
    for (int t = 0; t < 2; ++t) {
        int col = x0 + t * 16 + n;
        int pos = y * HW + col;
        float p = 0.f;
        v8us pk;
#pragma unroll
        for (int r = 0; r < 8; ++r) {
            int ch = ct * 16 + r + hi * 8;
            float gamma = accg[t][r] + bg[ch];
            float beta  = accb[t][r] + bbias[ch];
            float xv = x[((size_t)b * Cout + ch) * NPOS + pos];
            int g = ch >> 3;
            float m  = gmean[b * GRPS + g];
            float vv = gvar [b * GRPS + g];
            float xn = (xv - m) * rsqrtf(vv + EPSV);
            float hv = xn * (1.f + gamma) + beta;
            pk[r] = f2bf(hv);
            if (scoreAcc) p += hv * c2w[ch];
        }
        if (outP) {
            unsigned short* dst =
                outP + (((size_t)b * HP + (y + 1)) * HP + (col + 1)) * Cout + ct * 16 + hi * 8;
            *reinterpret_cast<v8us*>(dst) = pk;
        }
        if (scoreAcc) {
            p += __shfl_xor(p, 16, 32);
            if (hi == 0) atomicAdd(&scoreAcc[b * NTOK + pos], p);
        }
    }
}

// ---------------- score = sigmoid(acc + bias) ----------------
__global__ void score_finalize(const float* __restrict__ acc, const float* __restrict__ bptr,
                               float* __restrict__ score, int n) {
    int i = blockIdx.x * blockDim.x + threadIdx.x;
    if (i >= n) return;
    float v = acc[i] + bptr[0];
    score[i] = 1.f / (1.f + __expf(-v));
}

// ---------------- global min/max over all scores ----------------
__global__ void __launch_bounds__(1024)
minmax_kernel(const float* __restrict__ score, float* __restrict__ mm, int n) {
    float lo = 3.4e38f, hh = -3.4e38f;
    for (int i = threadIdx.x; i < n; i += 1024) { float v = score[i]; lo = fminf(lo, v); hh = fmaxf(hh, v); }
    __shared__ float smn[1024], smx[1024];
    smn[threadIdx.x] = lo; smx[threadIdx.x] = hh;
    __syncthreads();
    for (int o = 512; o > 0; o >>= 1) {
        if (threadIdx.x < o) {
            smn[threadIdx.x] = fminf(smn[threadIdx.x], smn[threadIdx.x + o]);
            smx[threadIdx.x] = fmaxf(smx[threadIdx.x], smx[threadIdx.x + o]);
        }
        __syncthreads();
    }
    if (threadIdx.x == 0) { mm[0] = smn[0]; mm[1] = smx[0]; }
}

// ---------------- per-batch descending bitonic argsort of 4096 scores in LDS ----------------
__global__ void __launch_bounds__(1024)
sort_kernel(const float* __restrict__ score, int* __restrict__ sortIdx,
            float* __restrict__ outTopk, float* __restrict__ outRemain,
            float* __restrict__ outScore, float* __restrict__ outMask,
            float* __restrict__ maskWs) {
    int b = blockIdx.x, tid = threadIdx.x;
    __shared__ float ky[NTOK];
    __shared__ int   ix[NTOK];
    for (int i = tid; i < NTOK; i += 1024) { ky[i] = score[b * NTOK + i]; ix[i] = i; }
    __syncthreads();
    for (int size = 2; size <= NTOK; size <<= 1) {
        for (int stride = size >> 1; stride > 0; stride >>= 1) {
            for (int t = tid; t < NTOK / 2; t += 1024) {
                int i = 2 * t - (t & (stride - 1));
                int j = i + stride;
                bool descBlk = ((i & size) == 0);
                float ki = ky[i], kj = ky[j];
                bool sw = descBlk ? (ki < kj) : (ki > kj);
                if (sw) { ky[i] = kj; ky[j] = ki; int t2 = ix[i]; ix[i] = ix[j]; ix[j] = t2; }
            }
            __syncthreads();
        }
    }
    for (int s = tid; s < NTOK; s += 1024) {
        int id = ix[s];
        sortIdx[b * NTOK + s] = id;
        float mv = (s < TOPK) ? 1.f : 0.f;
        maskWs [b * NTOK + id] = mv;
        outMask[b * NTOK + id] = mv;
        if (s < TOPK) { outTopk[b * TOPK + s] = (float)id; outScore[b * TOPK + s] = ky[s]; }
        else            outRemain[b * TOPK + (s - TOPK)] = (float)id;
    }
}

// ---------------- token layernorm + gather -> G[b][k][c] bf16 (channels contiguous) ------
__global__ void __launch_bounds__(32)
gather_kernel(const float* __restrict__ x, const float* __restrict__ score,
              const int* __restrict__ sortIdx, unsigned short* __restrict__ G) {
    int k = blockIdx.x, b = blockIdx.y, lane = threadIdx.x;
    int tok = sortIdx[b * NTOK + k];
    float v[8], s = 0.f, q = 0.f;
#pragma unroll
    for (int j = 0; j < 8; ++j) {
        int c = lane * 8 + j;                     // 8 consecutive channels per lane
        float t = x[((size_t)b * C_IN + c) * NPOS + tok];
        v[j] = t; s += t; q += t * t;
    }
#pragma unroll
    for (int o = 16; o > 0; o >>= 1) { s += __shfl_xor(s, o, 32); q += __shfl_xor(q, o, 32); }
    float m  = s * (1.f / 256.f);
    float rv = rsqrtf(q * (1.f / 256.f) - m * m + EPSV);
    float gs = score[b * NTOK + tok];
    v8us pk;
#pragma unroll
    for (int j = 0; j < 8; ++j) pk[j] = f2bf((v[j] - m) * rv * gs);
    *reinterpret_cast<v8us*>(&G[((size_t)b * TOPK + k) * C_IN + lane * 8]) = pk;
}

// ---------------- projection GEMM: sampled[b] = P(32x256) x G(256x2048) ----------------
__global__ void __launch_bounds__(32)
proj_wmma(const unsigned short* __restrict__ Ap,   // prepacked proj fragments
          const unsigned short* __restrict__ G,    // [B][2048][256] bf16, token-major
          float* __restrict__ out) {               // [B][32][2048]
    int mt = blockIdx.x, nt = blockIdx.y, b = blockIdx.z;
    int lane = threadIdx.x;
    int n  = lane & 15;
    int hi = lane >> 4;
    constexpr int NKT = C_IN / 32;                  // 8
    v8f acc = {};
    const unsigned short* abase = Ap + (((size_t)mt * NKT * 32) + lane) * 16;
    const unsigned short* gbase = G + ((size_t)b * TOPK + nt * 16 + n) * C_IN + hi * 16;
#pragma unroll
    for (int kt = 0; kt < NKT; ++kt) {
        v16bf a  = *reinterpret_cast<const v16bf*>(abase + kt * 512);
        v16bf fb = *reinterpret_cast<const v16bf*>(gbase + kt * 32);
        acc = __builtin_amdgcn_wmma_f32_16x16x32_bf16(false, a, false, fb, (short)0, acc, false, false);
    }
#pragma unroll
    for (int r = 0; r < 8; ++r) {
        int mrow = mt * 16 + r + hi * 8;
        out[((size_t)b * CODE + mrow) * TOPK + nt * 16 + n] = acc[r];
    }
}

// ---------------- 8x-upsampled binary / score maps ----------------
__global__ void maps_kernel(const float* __restrict__ mask, const float* __restrict__ score,
                            const float* __restrict__ mm,
                            float* __restrict__ outBin, float* __restrict__ outSm, int total) {
    int i = blockIdx.x * blockDim.x + threadIdx.x;
    if (i >= total) return;
    int b   = i >> 18;
    int rem = i & 262143;
    int yy  = rem >> 9, xx = rem & 511;
    int tok = (yy >> 3) * HW + (xx >> 3);
    outBin[i] = mask[b * NTOK + tok];
    float smin = mm[0], smax = mm[1];
    outSm[i] = (score[b * NTOK + tok] - smin) / fmaxf(smax - smin, 1e-5f);
}

// =======================================================================
extern "C" void kernel_launch(void* const* d_in, const int* in_sizes, int n_in,
                              void* d_out, int out_size, void* d_ws, size_t ws_size,
                              hipStream_t stream) {
    const float* x        = (const float*)d_in[0];
    const float* semantic = (const float*)d_in[1];
    const float* w_sh1 = (const float*)d_in[2];  const float* b_sh1 = (const float*)d_in[3];
    const float* w_g1  = (const float*)d_in[4];  const float* b_g1  = (const float*)d_in[5];
    const float* w_b1  = (const float*)d_in[6];  const float* b_b1  = (const float*)d_in[7];
    const float* c1w   = (const float*)d_in[8];  const float* c1b   = (const float*)d_in[9];
    const float* w_sh2 = (const float*)d_in[10]; const float* b_sh2 = (const float*)d_in[11];
    const float* w_g2  = (const float*)d_in[12]; const float* b_g2  = (const float*)d_in[13];
    const float* w_b2  = (const float*)d_in[14]; const float* b_b2  = (const float*)d_in[15];
    const float* c2w   = (const float*)d_in[16]; const float* c2b   = (const float*)d_in[17];
    const float* projw = (const float*)d_in[18];

    // ---- workspace carve-up ----
    size_t off = 0;
    auto carve = [&](size_t bytes) -> void* {
        void* p = (char*)d_ws + off;
        off += (bytes + 255) & ~(size_t)255;
        return p;
    };
    const int KpSH = 9 * 32;     // seg conv, channels padded 19->32
    const int KpGB = 9 * C_HID;  // 1152
    const int KpC1 = 9 * C_IN;   // 2304
    unsigned short* wsh1T = (unsigned short*)carve((size_t)KpSH * C_HID * 2);
    unsigned short* wsh2T = (unsigned short*)carve((size_t)KpSH * C_HID * 2);
    unsigned short* wg1T  = (unsigned short*)carve((size_t)KpGB * C_IN * 2);
    unsigned short* wb1T  = (unsigned short*)carve((size_t)KpGB * C_IN * 2);
    unsigned short* wg2T  = (unsigned short*)carve((size_t)KpGB * C_IN * 2);
    unsigned short* wb2T  = (unsigned short*)carve((size_t)KpGB * C_IN * 2);
    unsigned short* c1T   = (unsigned short*)carve((size_t)KpC1 * C_IN * 2);
    unsigned short* pT    = (unsigned short*)carve((size_t)C_IN * CODE * 2);
    unsigned short* segP  = (unsigned short*)carve((size_t)BATCH * HP * HP * 32 * 2);
    unsigned short* actvP = (unsigned short*)carve((size_t)BATCH * HP * HP * C_HID * 2);
    unsigned short* hs1P  = (unsigned short*)carve((size_t)BATCH * HP * HP * C_IN * 2);
    float* h1F     = (float*)carve((size_t)BATCH * C_IN * NPOS * 4);
    float* mean1   = (float*)carve(BATCH * GRPS * 4);
    float* var1    = (float*)carve(BATCH * GRPS * 4);
    float* mean2   = (float*)carve(BATCH * GRPS * 4);
    float* var2    = (float*)carve(BATCH * GRPS * 4);
    float* scoreAcc= (float*)carve((size_t)BATCH * NTOK * 4);
    float* pred    = (float*)carve((size_t)BATCH * NTOK * 4);
    int*   sortIdx = (int*)  carve((size_t)BATCH * NTOK * 4);
    float* maskWs  = (float*)carve((size_t)BATCH * NTOK * 4);
    float* mm      = (float*)carve(2 * 4);
    unsigned short* G = (unsigned short*)carve((size_t)BATCH * TOPK * C_IN * 2);

    // ---- output carve-up ----
    float* out       = (float*)d_out;
    float* oSampled  = out;
    float* oTopk     = oSampled + (size_t)BATCH * CODE * TOPK;
    float* oRemain   = oTopk    + (size_t)BATCH * TOPK;
    float* oBin      = oRemain  + (size_t)BATCH * TOPK;
    float* oSm       = oBin     + (size_t)BATCH * 512 * 512;
    float* oMask     = oSm      + (size_t)BATCH * 512 * 512;
    float* oScore    = oMask    + (size_t)BATCH * NTOK;

    // ---- 1) weight repack (tap-major fragment order) + padded NHWC segmap ----
    auto cvtW = [&](const float* w, unsigned short* wT, int Cout, int CinReal, int CINp, int ntaps) {
        int tot = ntaps * CINp * Cout;
        cvt_weight_frag<<<(tot + 255) / 256, 256, 0, stream>>>(w, wT, Cout, CinReal, CINp, ntaps);
    };
    cvtW(w_sh1, wsh1T, C_HID, LBL,   32,    9);
    cvtW(w_sh2, wsh2T, C_HID, LBL,   32,    9);
    cvtW(w_g1,  wg1T,  C_IN,  C_HID, C_HID, 9);
    cvtW(w_b1,  wb1T,  C_IN,  C_HID, C_HID, 9);
    cvtW(w_g2,  wg2T,  C_IN,  C_HID, C_HID, 9);
    cvtW(w_b2,  wb2T,  C_IN,  C_HID, C_HID, 9);
    cvtW(c1w,   c1T,   C_IN,  C_IN,  C_IN,  9);
    cvtW(projw, pT,    CODE,  C_IN,  C_IN,  1);
    {
        int n = BATCH * HP * HP * 32;
        cvt_seg_pad<<<(n + 255) / 256, 256, 0, stream>>>(semantic, segP);
    }
    // zero halos once per call; producers overwrite the interior
    hipMemsetAsync(actvP, 0, (size_t)BATCH * HP * HP * C_HID * 2, stream);
    hipMemsetAsync(hs1P,  0, (size_t)BATCH * HP * HP * C_IN  * 2, stream);

    // ---- 2) SPADE-1 ----
    gn_stats<<<dim3(BATCH, GRPS), 256, 0, stream>>>(x, mean1, var1);
    conv3x3_wmma<32><<<dim3(C_HID / 16, NPOS / 32, BATCH), 32, 0, stream>>>(
        segP, wsh1T, b_sh1, C_HID, actvP, nullptr, 1);
    spade_gb_apply<<<dim3(C_IN / 16, NPOS / 32, BATCH), 32, 0, stream>>>(
        actvP, wg1T, wb1T, b_g1, b_b1, x, mean1, var1, hs1P, nullptr, nullptr);

    // ---- 3) conv1 256->256 + relu (fp32 NCHW out for GN + SPADE-2 apply) ----
    conv3x3_wmma<C_IN><<<dim3(C_IN / 16, NPOS / 32, BATCH), 32, 0, stream>>>(
        hs1P, c1T, c1b, C_IN, nullptr, h1F, 1);

    // ---- 4) SPADE-2 + fused 1x1 score conv ----
    gn_stats<<<dim3(BATCH, GRPS), 256, 0, stream>>>(h1F, mean2, var2);
    conv3x3_wmma<32><<<dim3(C_HID / 16, NPOS / 32, BATCH), 32, 0, stream>>>(
        segP, wsh2T, b_sh2, C_HID, actvP, nullptr, 1);
    hipMemsetAsync(scoreAcc, 0, (size_t)BATCH * NTOK * 4, stream);
    spade_gb_apply<<<dim3(C_IN / 16, NPOS / 32, BATCH), 32, 0, stream>>>(
        actvP, wg2T, wb2T, b_g2, b_b2, h1F, mean2, var2, nullptr, scoreAcc, c2w);

    // ---- 5) sigmoid, min/max, argsort ----
    score_finalize<<<(BATCH * NTOK + 255) / 256, 256, 0, stream>>>(scoreAcc, c2b, pred, BATCH * NTOK);
    minmax_kernel<<<1, 1024, 0, stream>>>(pred, mm, BATCH * NTOK);
    sort_kernel<<<BATCH, 1024, 0, stream>>>(pred, sortIdx, oTopk, oRemain, oScore, oMask, maskWs);

    // ---- 6) token layernorm + gather, projection GEMM ----
    gather_kernel<<<dim3(TOPK, BATCH), 32, 0, stream>>>(x, pred, sortIdx, G);
    proj_wmma<<<dim3(CODE / 16, TOPK / 16, BATCH), 32, 0, stream>>>(pT, G, oSampled);

    // ---- 7) upsampled maps ----
    int totalMap = BATCH * 512 * 512;
    maps_kernel<<<(totalMap + 255) / 256, 256, 0, stream>>>(maskWs, pred, mm, oBin, oSm, totalMap);
}
